// TemporalAttn_43997644980904
// MI455X (gfx1250) — matrix operand order repaired and checked by reading, not compiled, not run on hardware
//
#include <hip/hip_runtime.h>
#include <hip/hip_bf16.h>

typedef _Float16 v16h __attribute__((ext_vector_type(16)));
typedef _Float16 v8h  __attribute__((ext_vector_type(8)));
typedef float    v8f  __attribute__((ext_vector_type(8)));
typedef float    v4f  __attribute__((ext_vector_type(4)));

union F16x16 { v16h v; v8h h[2]; };

#define B_    64
#define T_    256
#define D_    512
#define H_    8
#define HD_   64
#define S_    257
#define DFF_  2048
#define OUT_  512
#define MROWS (B_ * T_)   // 16384 rows of x viewed as (M, D)
#define MBLK  32
#define KCH   128

// ---------- split f32 -> f16 hi/lo planes (one-time, for weights) ----------
__global__ __launch_bounds__(256) void split_f32_kernel(
    const float* __restrict__ src, _Float16* __restrict__ hi,
    _Float16* __restrict__ lo, int n)
{
    int i = blockIdx.x * 256 + threadIdx.x;
    if (i < n) {
        float v = src[i];
        _Float16 h = (_Float16)v;
        hi[i] = h;
        lo[i] = (_Float16)(v - (float)h);
    }
}

// ---------- CLS projections: q0 = Wq@cls, kcls = Wk@cls, vcls = Wv@cls ----------
__global__ __launch_bounds__(256) void cls_proj_kernel(
    const float* __restrict__ cls, const float* __restrict__ Wq,
    const float* __restrict__ Wk, const float* __restrict__ Wv,
    float* __restrict__ q0, float* __restrict__ kcls, float* __restrict__ vcls)
{
    __shared__ float c[D_];
    for (int i = threadIdx.x; i < D_; i += 256) c[i] = cls[i];
    __syncthreads();
    for (int e = threadIdx.x; e < D_; e += 256) {
        float sq = 0.f, sk = 0.f, sv = 0.f;
        for (int d = 0; d < D_; ++d) {
            float cv = c[d];
            sq += Wq[(size_t)e * D_ + d] * cv;
            sk += Wk[(size_t)e * D_ + d] * cv;
            sv += Wv[(size_t)e * D_ + d] * cv;
        }
        q0[e] = sq; kcls[e] = sk; vcls[e] = sv;
    }
}

// ---------- K/V projection GEMM: split-f16 WMMA, fp32 accumulate ----------
// C[m,n] = sum_d x[m,d] * W[n,d].  blockIdx.y: 0 -> K proj, 1 -> V proj.
// WG = 256 threads (8 waves). WG tile: 32(M) x 512(N); wave tile: 32 x 64.
__global__ __launch_bounds__(256) void kv_gemm_kernel(
    const float* __restrict__ x,
    const _Float16* __restrict__ Wkhi, const _Float16* __restrict__ Wklo,
    const _Float16* __restrict__ Wvhi, const _Float16* __restrict__ Wvlo,
    float* __restrict__ Kmat, float* __restrict__ Vmat)
{
    __shared__ _Float16 Ahi[MBLK][KCH];
    __shared__ _Float16 Alo[MBLK][KCH];

    const int proj = blockIdx.y;
    const _Float16* __restrict__ Whi = proj ? Wvhi : Wkhi;
    const _Float16* __restrict__ Wlo = proj ? Wvlo : Wklo;
    float* __restrict__ Out = proj ? Vmat : Kmat;

    const int m0    = blockIdx.x * MBLK;
    const int tid   = threadIdx.x;
    const int wave  = tid >> 5;        // 0..7
    const int lane  = tid & 31;
    const int lcol  = lane & 15;       // tile row (A) / tile col (B,C)
    const int khalf = lane >> 4;       // 0 or 1
    const int n0    = wave * 64;       // wave's N base (4 n-tiles)

    v8f acc[2][4];
#pragma unroll
    for (int mt = 0; mt < 2; ++mt)
#pragma unroll
        for (int nt = 0; nt < 4; ++nt)
            acc[mt][nt] = (v8f){0.f, 0.f, 0.f, 0.f, 0.f, 0.f, 0.f, 0.f};

    for (int kc = 0; kc < D_; kc += KCH) {
        // ---- stage 32x128 A chunk into LDS with hi/lo split (once per WG) ----
        {
            int r  = tid >> 3;           // 0..31
            int c0 = (tid & 7) * 16;     // 0..112
            const float* src = x + (size_t)(m0 + r) * D_ + kc + c0;
#pragma unroll
            for (int i = 0; i < 16; i += 4) {
                v4f f = *(const v4f*)(src + i);
#pragma unroll
                for (int j = 0; j < 4; ++j) {
                    _Float16 hh = (_Float16)f[j];
                    Ahi[r][c0 + i + j] = hh;
                    Alo[r][c0 + i + j] = (_Float16)(f[j] - (float)hh);
                }
            }
        }
        __syncthreads();

#pragma unroll
        for (int kk = 0; kk < KCH; kk += 32) {
            // A fragments (16x32 f16 layout): lanes<16 -> K {0..7,16..23}, lanes>=16 -> +8
            F16x16 ahi[2], alo[2];
#pragma unroll
            for (int mt = 0; mt < 2; ++mt) {
                int row  = mt * 16 + lcol;
                int base = kk + 8 * khalf;
                ahi[mt].h[0] = *(const v8h*)&Ahi[row][base];
                ahi[mt].h[1] = *(const v8h*)&Ahi[row][base + 16];
                alo[mt].h[0] = *(const v8h*)&Alo[row][base];
                alo[mt].h[1] = *(const v8h*)&Alo[row][base + 16];
            }
            // B fragments: B[k][n] = W[n][k]; lanes<16 -> K {0..7,16..23}, lanes>=16 -> +8
            F16x16 bhi[4], blo[4];
#pragma unroll
            for (int nt = 0; nt < 4; ++nt) {
                int n = n0 + nt * 16 + lcol;
                size_t boff = (size_t)n * D_ + kc + kk + khalf * 8;
                bhi[nt].h[0] = *(const v8h*)(Whi + boff);
                bhi[nt].h[1] = *(const v8h*)(Whi + boff + 16);
                blo[nt].h[0] = *(const v8h*)(Wlo + boff);
                blo[nt].h[1] = *(const v8h*)(Wlo + boff + 16);
            }
            // split product: hi*hi + hi*lo + lo*hi (lo*lo dropped, ~2^-22 rel)
#pragma unroll
            for (int mt = 0; mt < 2; ++mt)
#pragma unroll
                for (int nt = 0; nt < 4; ++nt) {
                    acc[mt][nt] = __builtin_amdgcn_wmma_f32_16x16x32_f16(
                        false, ahi[mt].v, false, bhi[nt].v, (short)0, acc[mt][nt], false, false);
                    acc[mt][nt] = __builtin_amdgcn_wmma_f32_16x16x32_f16(
                        false, ahi[mt].v, false, blo[nt].v, (short)0, acc[mt][nt], false, false);
                    acc[mt][nt] = __builtin_amdgcn_wmma_f32_16x16x32_f16(
                        false, alo[mt].v, false, bhi[nt].v, (short)0, acc[mt][nt], false, false);
                }
        }
        __syncthreads();
    }

    // ---- writeback (C/D layout: VGPR i -> row i + 8*khalf, col = lane&15) ----
#pragma unroll
    for (int mt = 0; mt < 2; ++mt)
#pragma unroll
        for (int nt = 0; nt < 4; ++nt) {
            int col = n0 + nt * 16 + lcol;
#pragma unroll
            for (int i = 0; i < 8; ++i) {
                int row = m0 + mt * 16 + i + 8 * khalf;
                Out[(size_t)row * D_ + col] = acc[mt][nt][i];
            }
        }
}

// ---------- single-query attention per (b,h): softmax over 257 keys ----------
__global__ __launch_bounds__(256) void attn_kernel(
    const float* __restrict__ q0, const float* __restrict__ kcls,
    const float* __restrict__ vcls, const float* __restrict__ Kmat,
    const float* __restrict__ Vmat, float* __restrict__ res)
{
    __shared__ float q[HD_];
    __shared__ float attn[S_];
    __shared__ float red[256];
    __shared__ float pv[4][HD_];

    int b = blockIdx.x, h = blockIdx.y, t = threadIdx.x;
    if (t < HD_) q[t] = q0[h * HD_ + t];
    __syncthreads();

    float lmax = -3.0e38f;
    for (int s = t; s < S_; s += 256) {
        const float* kr = (s == 0) ? (kcls + h * HD_)
                                   : (Kmat + (size_t)(b * T_ + s - 1) * D_ + h * HD_);
        float dot = 0.f;
#pragma unroll
        for (int d = 0; d < HD_; ++d) dot += q[d] * kr[d];
        dot *= 0.125f;   // 1/sqrt(64)
        attn[s] = dot;
        lmax = fmaxf(lmax, dot);
    }
    red[t] = lmax;
    __syncthreads();
    for (int o = 128; o > 0; o >>= 1) { if (t < o) red[t] = fmaxf(red[t], red[t + o]); __syncthreads(); }
    float m = red[0];
    __syncthreads();

    float lsum = 0.f;
    for (int s = t; s < S_; s += 256) { float e = __expf(attn[s] - m); attn[s] = e; lsum += e; }
    red[t] = lsum;
    __syncthreads();
    for (int o = 128; o > 0; o >>= 1) { if (t < o) red[t] += red[t + o]; __syncthreads(); }
    float rz = 1.f / red[0];
    __syncthreads();

    int d = t & (HD_ - 1), part = t >> 6;
    float a = 0.f;
    for (int s = part; s < S_; s += 4) {
        const float* vr = (s == 0) ? (vcls + h * HD_)
                                   : (Vmat + (size_t)(b * T_ + s - 1) * D_ + h * HD_);
        a += attn[s] * vr[d];
    }
    pv[part][d] = a;
    __syncthreads();
    if (part == 0)
        res[(size_t)b * D_ + h * HD_ + d] = (pv[0][d] + pv[1][d] + pv[2][d] + pv[3][d]) * rz;
}

// ---------- head: cls_out[b,j] = sum_e Wo[j,e] * res[b,e] ----------
__global__ __launch_bounds__(256) void wo_kernel(
    const float* __restrict__ res, const float* __restrict__ Wo,
    float* __restrict__ clsout)
{
    int i = blockIdx.x * 256 + threadIdx.x;  // 64*64
    int b = i >> 6, j = i & 63;
    float s = 0.f;
    for (int e = 0; e < D_; ++e) s += Wo[(size_t)j * D_ + e] * res[(size_t)b * D_ + e];
    clsout[i] = s;
}

// ---------- hidden = relu(cls_out @ W1^T + b1) ----------
__global__ __launch_bounds__(256) void mlp1_kernel(
    const float* __restrict__ clsout, const float* __restrict__ W1,
    const float* __restrict__ b1, float* __restrict__ hidden)
{
    int i = blockIdx.x * 256 + threadIdx.x;  // 64*2048
    int b = i >> 11, f = i & 2047;
    float s = b1[f];
#pragma unroll
    for (int j = 0; j < HD_; ++j) s += W1[(size_t)f * HD_ + j] * clsout[b * HD_ + j];
    hidden[i] = fmaxf(s, 0.f);
}

// ---------- out = hidden @ W2^T + b2 ----------
__global__ __launch_bounds__(256) void mlp2_kernel(
    const float* __restrict__ hidden, const float* __restrict__ W2,
    const float* __restrict__ b2, float* __restrict__ out)
{
    int i = blockIdx.x * 256 + threadIdx.x;  // 64*512
    int b = i >> 9, o = i & 511;
    float s = b2[o];
    for (int f = 0; f < DFF_; ++f) s += W2[(size_t)o * DFF_ + f] * hidden[(size_t)b * DFF_ + f];
    out[i] = s;
}

extern "C" void kernel_launch(void* const* d_in, const int* in_sizes, int n_in,
                              void* d_out, int out_size, void* d_ws, size_t ws_size,
                              hipStream_t stream)
{
    const float* x   = (const float*)d_in[0];
    const float* cls = (const float*)d_in[1];
    const float* Wq  = (const float*)d_in[2];
    const float* Wk  = (const float*)d_in[3];
    const float* Wv  = (const float*)d_in[4];
    const float* Wo  = (const float*)d_in[5];
    const float* W1  = (const float*)d_in[6];
    const float* b1  = (const float*)d_in[7];
    const float* W2  = (const float*)d_in[8];
    const float* b2  = (const float*)d_in[9];
    float* out = (float*)d_out;

    // workspace carve-out (256B aligned); total ~67 MB
    char* base = (char*)d_ws;
    size_t off = 0;
    auto carve = [&](size_t bytes) -> void* {
        void* p = base + off;
        off = (off + bytes + 255) & ~(size_t)255;
        return p;
    };
    _Float16* wkhi = (_Float16*)carve((size_t)D_ * D_ * 2);
    _Float16* wklo = (_Float16*)carve((size_t)D_ * D_ * 2);
    _Float16* wvhi = (_Float16*)carve((size_t)D_ * D_ * 2);
    _Float16* wvlo = (_Float16*)carve((size_t)D_ * D_ * 2);
    float* q0     = (float*)carve((size_t)D_ * 4);
    float* kcls   = (float*)carve((size_t)D_ * 4);
    float* vcls   = (float*)carve((size_t)D_ * 4);
    float* res    = (float*)carve((size_t)B_ * D_ * 4);
    float* clsout = (float*)carve((size_t)B_ * HD_ * 4);
    float* hidden = (float*)carve((size_t)B_ * DFF_ * 4);
    float* Kmat   = (float*)carve((size_t)MROWS * D_ * 4);
    float* Vmat   = (float*)carve((size_t)MROWS * D_ * 4);

    // 1) split weights into f16 hi/lo planes (L2-resident thereafter)
    int nw = D_ * D_;
    split_f32_kernel<<<(nw + 255) / 256, 256, 0, stream>>>(Wk, wkhi, wklo, nw);
    split_f32_kernel<<<(nw + 255) / 256, 256, 0, stream>>>(Wv, wvhi, wvlo, nw);

    // 2) CLS projections (q0 shared across all batches)
    cls_proj_kernel<<<1, 256, 0, stream>>>(cls, Wq, Wk, Wv, q0, kcls, vcls);

    // 3) dominant GEMM: K and V projections via split-f16 WMMA
    kv_gemm_kernel<<<dim3(MROWS / MBLK, 2), 256, 0, stream>>>(
        x, wkhi, wklo, wvhi, wvlo, Kmat, Vmat);

    // 4) single-query attention per (b,h)
    attn_kernel<<<dim3(B_, H_), 256, 0, stream>>>(q0, kcls, vcls, Kmat, Vmat, res);

    // 5) output head + MLP
    wo_kernel<<<(B_ * HD_) / 256, 256, 0, stream>>>(res, Wo, clsout);
    mlp1_kernel<<<(B_ * DFF_) / 256, 256, 0, stream>>>(clsout, W1, b1, hidden);
    mlp2_kernel<<<(B_ * OUT_) / 256, 256, 0, stream>>>(hidden, W2, b2, out);
}
